// RNNModel_12936441496230
// MI455X (gfx1250) — matrix operand order repaired and checked by reading, not compiled, not run on hardware
//
#include <hip/hip_runtime.h>
#include <hip/hip_bf16.h>
#include <math.h>
#include <stdint.h>

// ---------------------------------------------------------------------------
// LN-LSTM (B=64, T=100, U=384, L=4) persistent-kernel implementation for
// gfx1250 (MI455X).  Design notes:
//  * GEMMs (M=64,N=1536,K=384) via v_wmma_f32_16x16x32_bf16, fp32 accum.
//  * Weights converted once to bf16 and TRANSPOSED to [N][K] so each lane's
//    B-fragment (N = lane%16, two 8-wide K runs at 8h and 16+8h) is two 16B contiguous reads.
//  * A-operand activations staged per layer-step into bank-padded LDS via
//    GLOBAL_LOAD_ASYNC_TO_LDS_B128 (ASYNCcnt) when the toolchain exposes it.
//  * 16 persistent workgroups, each owning a 96-column slice of N=1536.
//    Grid-wide sync via monotonic L2 atomic counter + agent-scope fences.
//  * LayerNorm reductions: per-WG partials written to fixed slots, summed in
//    fixed order by readers -> deterministic across graph replays.
// ---------------------------------------------------------------------------

typedef __bf16 bf16;
typedef __attribute__((ext_vector_type(16))) __bf16 v16bf;
typedef __attribute__((ext_vector_type(8)))  float  v8f;
typedef __attribute__((ext_vector_type(4)))  unsigned int v4u;
typedef __attribute__((ext_vector_type(4)))  int v4i;

union FragBF { v16bf v; v4u u[2]; };

static constexpr int Bn   = 64;     // batch
static constexpr int Tn   = 100;    // time steps
static constexpr int Un   = 384;    // hidden
static constexpr int NU   = 1536;   // 4*U (gates)
static constexpr int Ln   = 4;      // layers
static constexpr int NCLSn= 10;
static constexpr int NWG  = 16;     // persistent workgroups
static constexpr int NSL  = NU / NWG;  // 96  z-columns per WG
static constexpr int USL  = Un / NWG;  // 24  u-columns per WG
static constexpr int APAD = 392;    // LDS activation row stride in bf16
                                    // (196 DW == 4 mod 64 -> conflict-free b128)
static constexpr float EPSC = 1e-3f;

// ---- async global->LDS copy (gfx1250), with synchronous fallback ----------
#if defined(__has_builtin)
#if __has_builtin(__builtin_amdgcn_global_load_async_to_lds_b128) && \
    __has_builtin(__builtin_amdgcn_s_wait_asynccnt)
#define HAVE_ASYNC_LDS 1
#endif
#endif

#ifdef HAVE_ASYNC_LDS
typedef __attribute__((address_space(1))) v4i as1_v4i;
typedef __attribute__((address_space(3))) v4i as3_v4i;
// builtin signature (probed): (as1 v4i* gsrc, as3 v4i* ldst, imm offset, imm cpol)
// AS3 pointer is 32-bit; flat shared pointers carry the LDS byte offset in the
// low 32 bits, so the uintptr_t truncation yields the correct LDS address.
#define ACP128(ldst, gsrc)                                                    \
    __builtin_amdgcn_global_load_async_to_lds_b128(                           \
        reinterpret_cast<as1_v4i*>((uintptr_t)(gsrc)),                        \
        reinterpret_cast<as3_v4i*>((uintptr_t)(ldst)), 0, 0)
#define AWAIT() __builtin_amdgcn_s_wait_asynccnt(0)
#else
#define ACP128(ldst, gsrc) do { *(v4u*)(ldst) = *(const v4u*)(gsrc); } while (0)
#define AWAIT() do {} while (0)
#endif

// ---- workspace layout (bytes) ---------------------------------------------
static constexpr size_t OFF_BAR  = 0;                         // 1 uint, pad 256
static constexpr size_t OFF_HS   = 256;                       // h state f32 L*B*U
static constexpr size_t HS_B     = (size_t)Ln*Bn*Un*4;        // 393216
static constexpr size_t OFF_CS   = OFF_HS + HS_B;
static constexpr size_t OFF_HB   = OFF_CS + HS_B;             // h state bf16
static constexpr size_t HB_B     = (size_t)Ln*Bn*Un*2;        // 196608
static constexpr size_t FRONT_B  = OFF_HB + HB_B;             // memset region
static constexpr size_t OFF_WKT  = FRONT_B;                   // kernels^T bf16
static constexpr size_t WKT_B    = (size_t)(Ln-1)*NU*Un*2;    // 3538944
static constexpr size_t OFF_WRT  = OFF_WKT + WKT_B;           // rec^T bf16
static constexpr size_t WRT_B    = (size_t)Ln*NU*Un*2;        // 4718592
static constexpr size_t OFF_RED1 = OFF_WRT + WRT_B;           // per-iter,WG,row 4 stats
static constexpr size_t RED1_B   = (size_t)Tn*Ln*NWG*Bn*4*4;  // 6553600
static constexpr size_t OFF_RED2 = OFF_RED1 + RED1_B;         // per-iter,WG,row 2 stats
static constexpr size_t RED2_B   = (size_t)Tn*Ln*NWG*Bn*2*4;  // 3276800
static constexpr size_t OFF_ZA   = OFF_RED2 + RED2_B;         // raw kernel-path z
static constexpr size_t ZA_B     = (size_t)Bn*NU*4;           // 393216
static constexpr size_t OFF_ZR   = OFF_ZA + ZA_B;             // raw rec-path z
static constexpr size_t WS_TOTAL = OFF_ZR + ZA_B;             // ~19.86 MB

static constexpr size_t SLDS_B = (size_t)(2*Bn*APAD)*2                 // act bufs
                               + (size_t)(Bn*NSL*2 + Bn*USL*2 + Bn*4 + Bn*2)*4;

__device__ __forceinline__ float sigmoidf_(float x) { return 1.0f / (1.0f + expf(-x)); }

__device__ __forceinline__ void gsync(unsigned* bar, unsigned& target) {
    __syncthreads();
    __builtin_amdgcn_fence(__ATOMIC_RELEASE, "agent");
    if (threadIdx.x == 0) {
        __hip_atomic_fetch_add(bar, 1u, __ATOMIC_RELAXED, __HIP_MEMORY_SCOPE_AGENT);
        while (__hip_atomic_load(bar, __ATOMIC_RELAXED, __HIP_MEMORY_SCOPE_AGENT) < target) {
            __builtin_amdgcn_s_sleep(2);
        }
    }
    target += NWG;
    __syncthreads();
    __builtin_amdgcn_fence(__ATOMIC_ACQUIRE, "agent");
}

// ---- one-time weight convert + transpose: [l][k][n] f32 -> [l][n][k] bf16 --
extern "C" __global__ __launch_bounds__(256)
void prep_weights(const float* __restrict__ kern, const float* __restrict__ reck,
                  bf16* __restrict__ WkT, bf16* __restrict__ WrT) {
    const long PER = (long)NU * Un;
    long i = (long)blockIdx.x * 256 + threadIdx.x;
    if (i < (long)Ln * PER) {
        long l = i / PER, r = i % PER;
        long n = r / Un, k = r % Un;
        WrT[i] = (bf16)reck[l * PER + k * NU + n];
    }
    if (i < (long)(Ln - 1) * PER) {
        long l = i / PER, r = i % PER;
        long n = r / Un, k = r % Un;
        WkT[i] = (bf16)kern[l * PER + k * NU + n];
    }
}

// ---- main persistent kernel -----------------------------------------------
extern "C" __global__ __launch_bounds__(256, 1)
void lstm_persistent(const float* __restrict__ x,
                     const unsigned char* __restrict__ mask,  // jnp bool -> 1 byte
                     const float* __restrict__ k0,
                     const float* __restrict__ biases,
                     const float* __restrict__ kn_g, const float* __restrict__ kn_b,
                     const float* __restrict__ rn_g, const float* __restrict__ rn_b,
                     const float* __restrict__ sn_g, const float* __restrict__ sn_b,
                     const float* __restrict__ dense_w, const float* __restrict__ dense_b,
                     unsigned* __restrict__ bar,
                     float* __restrict__ red1, float* __restrict__ red2,
                     float* __restrict__ hS, float* __restrict__ cS, bf16* __restrict__ hB,
                     const bf16* __restrict__ WkT, const bf16* __restrict__ WrT,
                     float* __restrict__ zA, float* __restrict__ zR,
                     float* __restrict__ out) {
    extern __shared__ char smem[];
    bf16*  sAct0 = (bf16*)smem;              // [64][392] kernel-path A (layer input)
    bf16*  sAct1 = sAct0 + Bn * APAD;        // [64][392] rec-path A (h state)
    float* sZA   = (float*)(sAct1 + Bn * APAD); // [64][96] raw kernel-path z slice
    float* sZR   = sZA + Bn * NSL;           // [64][96] raw rec-path z slice
    float* sCP   = sZR + Bn * NSL;           // [64][24] pre-LN cell values
    float* sZO   = sCP + Bn * USL;           // [64][24] output-gate pre-acts
    float* sST   = sZO + Bn * USL;           // [64][4]  {mA,rA,mR,rR}
    float* sST2  = sST + Bn * 4;             // [64][2]  {m,rstd}

    const int tid  = threadIdx.x;
    const int lane = tid & 31;
    const int wave = tid >> 5;
    const int g    = blockIdx.x;
    unsigned target = NWG;

    for (int t = 0; t < Tn; ++t) {
        for (int l = 0; l < Ln; ++l) {
            const int iter = t * Ln + l;

            // ---- stage A-operand activations into padded LDS (async) ------
            {
                bf16* gsrc1 = hB + (size_t)l * Bn * Un;
                bf16* gsrc0 = hB + (size_t)(l > 0 ? l - 1 : 0) * Bn * Un;
                for (int ch = tid; ch < Bn * (Un / 8); ch += 256) {  // 16B chunks
                    const int r = ch / (Un / 8), c = ch % (Un / 8);
                    const int  loff = r * APAD + c * 8;
                    const size_t goff = (size_t)r * Un + c * 8;
                    ACP128(sAct1 + loff, gsrc1 + goff);
                    if (l > 0) ACP128(sAct0 + loff, gsrc0 + goff);
                }
                AWAIT();
            }
            __syncthreads();

            // ================= Phase G: GEMMs into the WG's N-slice ========
            const int ntiles = (l == 0) ? 24 : 48;   // l==0: kernel path is K=1 scalar
            for (int tile = wave; tile < ntiles; tile += 8) {
                const int p  = (l == 0) ? 1 : (tile / 24);  // 0: kernel path, 1: rec path
                const int q  = tile % 24;
                const int mt = q / 6, nt = q % 6;
                const bf16* act = p ? sAct1 : sAct0;
                const bf16* wt  = p ? (WrT + (size_t)l * NU * Un)
                                    : (WkT + (size_t)(l - 1) * NU * Un);
                const int m0 = mt * 16;
                const int n0 = g * NSL + nt * 16;
                // A fragment: row = m0+lane%16; K chunks [kb,kb+8) and [kb+16,kb+24)
                const bf16* aptr = act + (m0 + (lane & 15)) * APAD + ((lane >> 4) * 8);
                // B fragment (from [N][K] bf16): N = n0+lane%16; K chunks [kb,kb+8) and [kb+16,kb+24)
                const bf16* bptr = wt  + (size_t)(n0 + (lane & 15)) * Un + ((lane >> 4) * 8);
                v8f acc = {};
#pragma unroll
                for (int kt = 0; kt < 12; ++kt) {
                    FragBF fa, fb;
                    fa.u[0] = *(const v4u*)(aptr + kt * 32);        // ds_load_b128
                    fa.u[1] = *(const v4u*)(aptr + kt * 32 + 16);   // ds_load_b128
                    fb.u[0] = *(const v4u*)(bptr + kt * 32);        // global_load_b128
                    fb.u[1] = *(const v4u*)(bptr + kt * 32 + 16);   // global_load_b128
                    if (kt + 1 < 12) __builtin_prefetch((const void*)(bptr + (kt + 1) * 32), 0, 3);
                    acc = __builtin_amdgcn_wmma_f32_16x16x32_bf16(
                        false, fa.v, false, fb.v, (short)0, acc, false, false);
                }
                float* sZ = p ? sZR : sZA;
                const int col = nt * 16 + (lane & 15);
                const int r0  = m0 + (lane >> 4) * 8;
#pragma unroll
                for (int i = 0; i < 8; ++i) sZ[(r0 + i) * NSL + col] = acc[i];
            }
            if (l == 0) {  // kernel path is x_t[b] * k0[n] (K == 1)
                for (int e = tid; e < Bn * NSL; e += 256) {
                    const int b = e / NSL, c = e % NSL;
                    sZA[e] = x[b * Tn + t] * k0[g * NSL + c];
                }
            }
            __syncthreads();
            // write slice to global (gate phase needs cross-WG columns)
            for (int e = tid; e < Bn * NSL; e += 256) {
                const int b = e / NSL, c = e % NSL;
                zA[(size_t)b * NU + g * NSL + c] = sZA[e];
                zR[(size_t)b * NU + g * NSL + c] = sZR[e];
            }
            // deterministic per-row partial stats for the two 1536-wide LNs
            if (tid < Bn) {
                float sA = 0, qA = 0, sR = 0, qR = 0;
                for (int c = 0; c < NSL; ++c) {
                    const float a = sZA[tid * NSL + c];
                    const float r = sZR[tid * NSL + c];
                    sA += a; qA += a * a; sR += r; qR += r * r;
                }
                float* dst = red1 + (((size_t)iter * NWG + g) * Bn + tid) * 4;
                dst[0] = sA; dst[1] = qA; dst[2] = sR; dst[3] = qR;
            }
            gsync(bar, target);

            // ================= Phase GATE: LN(z) + gates + c-pre ===========
            if (tid < Bn) {  // fixed-order sum of the 16 WG partials
                float sA = 0, qA = 0, sR = 0, qR = 0;
                const float* base = red1 + (size_t)iter * NWG * Bn * 4 + (size_t)tid * 4;
                for (int w = 0; w < NWG; ++w) {
                    const float* pp = base + (size_t)w * Bn * 4;
                    sA += pp[0]; qA += pp[1]; sR += pp[2]; qR += pp[3];
                }
                const float inv = 1.0f / (float)NU;
                const float mA = sA * inv, mR = sR * inv;
                sST[tid * 4 + 0] = mA;
                sST[tid * 4 + 1] = rsqrtf(fmaxf(qA * inv - mA * mA, 0.f) + EPSC);
                sST[tid * 4 + 2] = mR;
                sST[tid * 4 + 3] = rsqrtf(fmaxf(qR * inv - mR * mR, 0.f) + EPSC);
            }
            __syncthreads();
            for (int e = tid; e < Bn * USL; e += 256) {
                const int b = e / USL, uu = e % USL;
                const int u = g * USL + uu;
                const float mA = sST[b * 4 + 0], rA = sST[b * 4 + 1];
                const float mR = sST[b * 4 + 2], rR = sST[b * 4 + 3];
                float zv[4];
#pragma unroll
                for (int gi = 0; gi < 4; ++gi) {
                    const int col = u + gi * Un;
                    const float a = zA[(size_t)b * NU + col];
                    const float r = zR[(size_t)b * NU + col];
                    zv[gi] = (a - mA) * rA * kn_g[l * NU + col] + kn_b[l * NU + col]
                           + (r - mR) * rR * rn_g[l * NU + col] + rn_b[l * NU + col]
                           + biases[l * NU + col];
                }
                const float ci = sigmoidf_(zv[0]);
                const float cf = sigmoidf_(zv[1]);
                const float cg = tanhf(zv[2]);
                const float cold = cS[((size_t)l * Bn + b) * Un + u];
                const float cp = cf * cold + ci * cg;
                sCP[b * USL + uu] = cp;
                sZO[b * USL + uu] = zv[3];
            }
            __syncthreads();
            if (tid < Bn) {  // per-WG partial stats for the 384-wide cell LN
                float s = 0, qq = 0;
                for (int c = 0; c < USL; ++c) {
                    const float v = sCP[tid * USL + c];
                    s += v; qq += v * v;
                }
                float* dst = red2 + (((size_t)iter * NWG + g) * Bn + tid) * 2;
                dst[0] = s; dst[1] = qq;
            }
            gsync(bar, target);

            // ================= Phase OUT: LN(c), h update, mask ============
            if (tid < Bn) {
                float s = 0, qq = 0;
                const float* base = red2 + (size_t)iter * NWG * Bn * 2 + (size_t)tid * 2;
                for (int w = 0; w < NWG; ++w) {
                    const float* pp = base + (size_t)w * Bn * 2;
                    s += pp[0]; qq += pp[1];
                }
                const float inv = 1.0f / (float)Un;
                const float m = s * inv;
                sST2[tid * 2 + 0] = m;
                sST2[tid * 2 + 1] = rsqrtf(fmaxf(qq * inv - m * m, 0.f) + EPSC);
            }
            __syncthreads();
            for (int e = tid; e < Bn * USL; e += 256) {
                const int b = e / USL, uu = e % USL;
                const int u = g * USL + uu;
                const float m  = sST2[b * 2 + 0];
                const float rs = sST2[b * 2 + 1];
                const float cp = sCP[b * USL + uu];
                const float cn = (cp - m) * rs * sn_g[l * Un + u] + sn_b[l * Un + u];
                const float hn = sigmoidf_(sZO[b * USL + uu]) * tanhf(cn);
                const bool mk = mask[b * Tn + t] != 0;
                const size_t si = ((size_t)l * Bn + b) * Un + u;
                const float hv = mk ? hn : hS[si];
                const float cv = mk ? cn : cS[si];
                hS[si] = hv; cS[si] = cv; hB[si] = (bf16)hv;
            }
            gsync(bar, target);
        }
    }

    // ================= final dense (64x384 @ 384x10) + softmax =============
    if (blockIdx.x == 0 && tid < Bn) {
        float logit[NCLSn];
#pragma unroll
        for (int c = 0; c < NCLSn; ++c) logit[c] = dense_b[c];
        const float* hrow = hS + ((size_t)(Ln - 1) * Bn + tid) * Un;
        for (int u = 0; u < Un; ++u) {
            const float h = hrow[u];
#pragma unroll
            for (int c = 0; c < NCLSn; ++c) logit[c] += h * dense_w[u * NCLSn + c];
        }
        float mx = logit[0];
#pragma unroll
        for (int c = 1; c < NCLSn; ++c) mx = fmaxf(mx, logit[c]);
        float s = 0.f;
#pragma unroll
        for (int c = 0; c < NCLSn; ++c) { logit[c] = expf(logit[c] - mx); s += logit[c]; }
        const float is = 1.0f / s;
#pragma unroll
        for (int c = 0; c < NCLSn; ++c) out[tid * NCLSn + c] = logit[c] * is;
    }
}

// ---------------------------------------------------------------------------
extern "C" void kernel_launch(void* const* d_in, const int* in_sizes, int n_in,
                              void* d_out, int out_size, void* d_ws, size_t ws_size,
                              hipStream_t stream) {
    (void)in_sizes; (void)n_in; (void)out_size; (void)ws_size;
    const float* x    = (const float*)d_in[0];
    const unsigned char* mask = (const unsigned char*)d_in[1];
    const float* k0   = (const float*)d_in[2];
    const float* kern = (const float*)d_in[3];
    const float* reck = (const float*)d_in[4];
    const float* bia  = (const float*)d_in[5];
    const float* kng  = (const float*)d_in[6];
    const float* knb  = (const float*)d_in[7];
    const float* rng  = (const float*)d_in[8];
    const float* rnb  = (const float*)d_in[9];
    const float* sng  = (const float*)d_in[10];
    const float* snb  = (const float*)d_in[11];
    const float* dw   = (const float*)d_in[12];
    const float* db   = (const float*)d_in[13];
    float* out = (float*)d_out;

    char* ws = (char*)d_ws;
    unsigned* bar = (unsigned*)(ws + OFF_BAR);
    float* hS   = (float*)(ws + OFF_HS);
    float* cS   = (float*)(ws + OFF_CS);
    bf16*  hB   = (bf16*) (ws + OFF_HB);
    bf16*  WkT  = (bf16*) (ws + OFF_WKT);
    bf16*  WrT  = (bf16*) (ws + OFF_WRT);
    float* red1 = (float*)(ws + OFF_RED1);
    float* red2 = (float*)(ws + OFF_RED2);
    float* zA   = (float*)(ws + OFF_ZA);
    float* zR   = (float*)(ws + OFF_ZR);

    // zero barrier + h/c state each launch (graph-capturable memset node)
    (void)hipMemsetAsync(ws, 0, FRONT_B, stream);

    const long PER = (long)NU * Un;
    const unsigned prep_blocks = (unsigned)((Ln * PER + 255) / 256);
    prep_weights<<<prep_blocks, 256, 0, stream>>>(kern, reck, WkT, WrT);

    lstm_persistent<<<NWG, 256, SLDS_B, stream>>>(
        x, mask, k0, bia, kng, knb, rng, rnb, sng, snb, dw, db,
        bar, red1, red2, hS, cS, hB, WkT, WrT, zA, zR, out);
}
